// Model_19361712570450
// MI455X (gfx1250) — compile-verified, matrix operand order checked by reading, not run on hardware
//
#include <hip/hip_runtime.h>
#include <hip/hip_bf16.h>
#include <math.h>

// Problem dims
#define B_   4
#define S_   1024
#define E_   256
#define H_   256
#define G_   1024          // 4H
#define V_   32000
#define M_   (B_ * S_)     // 4096 token rows
#define K2_  512           // 2H

typedef __attribute__((ext_vector_type(16))) _Float16 v16h;
typedef __attribute__((ext_vector_type(8)))  float    v8f;
typedef __attribute__((ext_vector_type(4)))  int      v4i;

union Frag16 { v16h v; uint4 q[2]; };

// Load a 16x32 f16 A-fragment (or 32x16 B-fragment: B[k][n]=W[n][k], rows=n)
// per ISA 7.12.2: lane l -> row (l&15), two contiguous 8-element K chunks at
// kbase=(l>>4)*8 and kbase+16. ld = row pitch in elements (multiple of 8).
__device__ __forceinline__ v16h load_frag(const _Float16* p, int ld, int lane) {
  const int r  = lane & 15;
  const int kb = (lane >> 4) << 3;
  Frag16 u;
  u.q[0] = *(const uint4*)(p + (size_t)r * ld + kb);
  u.q[1] = *(const uint4*)(p + (size_t)r * ld + kb + 16);
  return u.v;
}

__device__ __forceinline__ float sigm(float x) { return 1.0f / (1.0f + __expf(-x)); }

// ---- gfx1250 async global->LDS copy (ASYNCcnt-tracked), 16B per lane ----
#define GAS1 __attribute__((address_space(1)))
#define LAS3 __attribute__((address_space(3)))

__device__ __forceinline__ void async_cp16(const _Float16* g, _Float16* l) {
#if __has_builtin(__builtin_amdgcn_global_load_async_to_lds_b128)
  __builtin_amdgcn_global_load_async_to_lds_b128(
      (GAS1 v4i*)g, (LAS3 v4i*)l, 0, 0);
#else
  // vdst = LDS byte address (low 32 bits of generic shared address), vaddr = 64-bit global
  asm volatile("global_load_async_to_lds_b128 %0, %1, off"
               :: "v"((unsigned)(unsigned long long)l), "v"(g) : "memory");
#endif
}

__device__ __forceinline__ void wait_async0() {
#if __has_builtin(__builtin_amdgcn_s_wait_asynccnt)
  __builtin_amdgcn_s_wait_asynccnt(0);
#else
  asm volatile("s_wait_asynccnt 0" ::: "memory");
#endif
}

// ---------------- prep kernels ----------------

__global__ void k_cvt_f16(const float* __restrict__ src, _Float16* __restrict__ dst, int n) {
  int i = blockIdx.x * blockDim.x + threadIdx.x;
  int stride = gridDim.x * blockDim.x;
  for (; i < n; i += stride) dst[i] = (_Float16)src[i];
}

__global__ void k_embed(const int* __restrict__ tok, const float* __restrict__ tab,
                        _Float16* __restrict__ emb) {
  int i = blockIdx.x * blockDim.x + threadIdx.x;
  int stride = gridDim.x * blockDim.x;
  for (; i < M_ * E_; i += stride) {
    int m = i >> 8, e = i & (E_ - 1);
    emb[i] = (_Float16)tab[(size_t)tok[m] * E_ + e];
  }
}

__global__ void k_bias(const float* __restrict__ bfi, const float* __restrict__ bfh,
                       const float* __restrict__ bbi, const float* __restrict__ bbh,
                       float* __restrict__ comb) {
  int i = blockIdx.x * blockDim.x + threadIdx.x;
  if (i < G_)            comb[i] = bfi[i] + bfh[i];
  else if (i < 2 * G_)   comb[i] = bbi[i - G_] + bbh[i - G_];
}

// Pre-swizzle Whh into per-lane-contiguous B fragments:
// packed[(((d*64+nt)*8+ks)*32+lane)*16 + j] = f16(Whh_d[n*H + k]) with
// n = nt*16 + (lane&15), k = ks*32 + klocal(j,lane).
__global__ void k_pack_whh(const float* __restrict__ whh_f, const float* __restrict__ whh_b,
                           _Float16* __restrict__ packed) {
  int i = blockIdx.x * blockDim.x + threadIdx.x;
  int stride = gridDim.x * blockDim.x;
  for (; i < 2 * G_ * H_; i += stride) {
    int d    = i >> 18;
    int r    = i & ((1 << 18) - 1);
    int j    = r & 15;
    int lane = (r >> 4) & 31;
    int ks   = (r >> 9) & 7;
    int nt   = r >> 12;
    int vreg = j >> 1, p = j & 1;
    int klocal = ((vreg & 3) << 1) + p + ((vreg >> 2) << 4) + ((lane >> 4) << 3);
    int n = (nt << 4) + (lane & 15);
    int k = (ks << 5) + klocal;
    const float* w = d ? whh_b : whh_f;
    packed[i] = (_Float16)w[(size_t)n * H_ + k];
  }
}

// ---------------- input projection: pre = emb @ Wih^T + (bih+bhh) ----------------
// B strip (64 gates x K=256, 32KB contiguous) staged to LDS via async copies.
__global__ __launch_bounds__(256) void k_input_gemm(
    const _Float16* __restrict__ emb, const _Float16* __restrict__ wih_all,
    const float* __restrict__ bias_comb, float* __restrict__ pre_all) {
  __shared__ __attribute__((aligned(16))) _Float16 bsh[64 * E_];   // 32 KB

  const int tid  = threadIdx.x;
  const int lane = tid & 31;
  const int wave = tid >> 5;
  const int dir  = blockIdx.z;
  const _Float16* wih = wih_all + (size_t)dir * G_ * E_;
  const float* bias   = bias_comb + dir * G_;
  float* pre          = pre_all + (size_t)dir * S_ * B_ * G_;

  const int m0 = (blockIdx.x * 8 + wave) << 4;   // 16-row M tile
  const int n0 = blockIdx.y << 6;                // 64-col N strip

  {
    const _Float16* bsrc = wih + (size_t)n0 * E_;          // 64 contiguous rows
    for (int c = tid; c < (64 * E_) / 8; c += 256)          // 16B chunks
      async_cp16(bsrc + (size_t)c * 8, &bsh[c * 8]);
    wait_async0();
  }
  __syncthreads();

  v8f z = {};
  v8f acc[4] = {z, z, z, z};

  for (int ks = 0; ks < 8; ++ks) {               // K = 256
    v16h a = load_frag(emb + (size_t)m0 * E_ + (ks << 5), E_, lane);
#pragma unroll
    for (int t = 0; t < 4; ++t) {
      v16h b = load_frag(&bsh[(size_t)(t << 4) * E_ + (ks << 5)], E_, lane);
      acc[t] = __builtin_amdgcn_wmma_f32_16x16x32_f16(false, a, false, b,
                                                      (short)0, acc[t], false, false);
    }
  }

  const int mrow = (lane >> 4) << 3;
  const int col  = lane & 15;
#pragma unroll
  for (int t = 0; t < 4; ++t) {
    int g = n0 + (t << 4) + col;
    float bv = bias[g];
#pragma unroll
    for (int r = 0; r < 8; ++r) {
      int m = m0 + mrow + r;
      int b = m >> 10, s = m & (S_ - 1);
      int tt = dir ? (S_ - 1 - s) : s;
      pre[((size_t)tt * B_ + b) * G_ + g] = acc[t][r] + bv;
    }
  }
}

// ---------------- sequential LSTM scan (one workgroup per direction) ----------------
__global__ __launch_bounds__(1024) void k_scan(
    const _Float16* __restrict__ whh_packed_all, const float* __restrict__ pre_all,
    _Float16* __restrict__ hcat) {
  const int dir  = blockIdx.x;
  const int tid  = threadIdx.x;
  const int lane = tid & 31;
  const int wave = tid >> 5;
  const _Float16* wp = whh_packed_all + (size_t)dir * G_ * H_;
  const float*   pre = pre_all + (size_t)dir * S_ * B_ * G_;

  __shared__ __attribute__((aligned(16))) _Float16 h_lds[16][H_]; // A operand, rows 4..15 = 0
  __shared__ float c_state[B_][H_];
  __shared__ float gates[B_][G_];

  for (int i = tid; i < 16 * H_; i += 1024) (&h_lds[0][0])[i] = (_Float16)0.0f;
  for (int i = tid; i < B_ * H_; i += 1024) (&c_state[0][0])[i] = 0.0f;
  __syncthreads();

  const int nt0   = wave << 1;      // each wave: two 16-col gate tiles
  const int b_act = tid >> 8;
  const int j_act = tid & (H_ - 1);

  for (int t = 0; t < S_; ++t) {
    v8f z = {};
    v8f acc0 = z, acc1 = z;
#pragma unroll
    for (int ks = 0; ks < 8; ++ks) {            // K = H = 256
      v16h a = load_frag((const _Float16*)&h_lds[0][0] + (ks << 5), H_, lane);
      Frag16 u0, u1;
      const uint4* q0 = (const uint4*)(wp + ((size_t)(nt0 * 8 + ks) * 32 + lane) * 16);
      u0.q[0] = q0[0]; u0.q[1] = q0[1];
      acc0 = __builtin_amdgcn_wmma_f32_16x16x32_f16(false, a, false, u0.v,
                                                    (short)0, acc0, false, false);
      const uint4* q1 = (const uint4*)(wp + ((size_t)((nt0 + 1) * 8 + ks) * 32 + lane) * 16);
      u1.q[0] = q1[0]; u1.q[1] = q1[1];
      acc1 = __builtin_amdgcn_wmma_f32_16x16x32_f16(false, a, false, u1.v,
                                                    (short)0, acc1, false, false);
    }
    if (lane < 16) {                             // only batch rows 0..3 are live
#pragma unroll
      for (int r = 0; r < 4; ++r) {
        gates[r][(nt0 << 4) + lane]       = acc0[r];
        gates[r][((nt0 + 1) << 4) + lane] = acc1[r];
      }
    }
    __syncthreads();
    {
      const float* p = pre + ((size_t)t * B_ + b_act) * G_;
      float gi = gates[b_act][j_act]            + p[j_act];
      float gf = gates[b_act][H_ + j_act]       + p[H_ + j_act];
      float gg = gates[b_act][2 * H_ + j_act]   + p[2 * H_ + j_act];
      float go = gates[b_act][3 * H_ + j_act]   + p[3 * H_ + j_act];
      float c = sigm(gf) * c_state[b_act][j_act] + sigm(gi) * tanhf(gg);
      float h = sigm(go) * tanhf(c);
      c_state[b_act][j_act] = c;
      h_lds[b_act][j_act]   = (_Float16)h;
      int s_out = dir ? (S_ - 1 - t) : t;
      hcat[((size_t)(b_act << 10) + s_out) * K2_ + (dir << 8) + j_act] = (_Float16)h;
    }
    __syncthreads();
  }
}

// ---------------- output projection: logits = hcat @ Wl^T + bl ----------------
// B strip (64 vocab rows x K=512, 64KB contiguous) staged to LDS via async copies.
__global__ __launch_bounds__(256) void k_final_gemm(
    const _Float16* __restrict__ hcat, const _Float16* __restrict__ wl,
    const float* __restrict__ bl, float* __restrict__ out) {
  __shared__ __attribute__((aligned(16))) _Float16 bsh[64 * K2_];  // 64 KB

  const int tid  = threadIdx.x;
  const int lane = tid & 31;
  const int wave = tid >> 5;
  const int m0 = (blockIdx.x * 8 + wave) << 4;   // M tile
  const int n0 = blockIdx.y << 6;                // 64-col N strip

  {
    const _Float16* bsrc = wl + (size_t)n0 * K2_;           // 64 contiguous rows
    for (int c = tid; c < (64 * K2_) / 8; c += 256)          // 16B chunks
      async_cp16(bsrc + (size_t)c * 8, &bsh[c * 8]);
    wait_async0();
  }
  __syncthreads();

  v8f z = {};
  v8f acc[4] = {z, z, z, z};

  for (int ks = 0; ks < 16; ++ks) {              // K = 512
    v16h a = load_frag(hcat + (size_t)m0 * K2_ + (ks << 5), K2_, lane);
#pragma unroll
    for (int t = 0; t < 4; ++t) {
      v16h b = load_frag(&bsh[(size_t)(t << 4) * K2_ + (ks << 5)], K2_, lane);
      acc[t] = __builtin_amdgcn_wmma_f32_16x16x32_f16(false, a, false, b,
                                                      (short)0, acc[t], false, false);
    }
  }

  const int mrow = (lane >> 4) << 3;
  const int col  = lane & 15;
#pragma unroll
  for (int t = 0; t < 4; ++t) {
    int n = n0 + (t << 4) + col;
    float bv = bl[n];
#pragma unroll
    for (int r = 0; r < 8; ++r) {
      int m = m0 + mrow + r;
      out[(size_t)m * V_ + n] = acc[t][r] + bv;   // flat [B,S,V] == reference's raw reshape
    }
  }
}

// ---------------- host ----------------
extern "C" void kernel_launch(void* const* d_in, const int* in_sizes, int n_in,
                              void* d_out, int out_size, void* d_ws, size_t ws_size,
                              hipStream_t stream) {
  (void)in_sizes; (void)n_in; (void)out_size; (void)ws_size;
  const int*   tokens = (const int*)  d_in[0];
  const float* table  = (const float*)d_in[1];
  const float* Wih_f  = (const float*)d_in[2];
  const float* Whh_f  = (const float*)d_in[3];
  const float* bih_f  = (const float*)d_in[4];
  const float* bhh_f  = (const float*)d_in[5];
  const float* Wih_b  = (const float*)d_in[6];
  const float* Whh_b  = (const float*)d_in[7];
  const float* bih_b  = (const float*)d_in[8];
  const float* bhh_b  = (const float*)d_in[9];
  const float* Wl     = (const float*)d_in[10];
  const float* bl     = (const float*)d_in[11];
  float* out = (float*)d_out;

  char* ws = (char*)d_ws;
  size_t o = 0;
  _Float16* emb16   = (_Float16*)(ws + o); o += (size_t)M_ * E_ * 2;          // 2 MB
  _Float16* wih16   = (_Float16*)(ws + o); o += (size_t)2 * G_ * E_ * 2;      // 1 MB
  _Float16* whh_pk  = (_Float16*)(ws + o); o += (size_t)2 * G_ * H_ * 2;      // 1 MB
  _Float16* wl16    = (_Float16*)(ws + o); o += (size_t)V_ * K2_ * 2;         // 32 MB
  float*    biasc   = (float*)   (ws + o); o += (size_t)2 * G_ * 4;           // 8 KB
  float*    pre     = (float*)   (ws + o); o += (size_t)2 * S_ * B_ * G_ * 4; // 32 MB
  _Float16* hcat    = (_Float16*)(ws + o); o += (size_t)M_ * K2_ * 2;         // 4 MB

  k_embed   <<<1024, 256, 0, stream>>>(tokens, table, emb16);
  k_cvt_f16 <<<512,  256, 0, stream>>>(Wih_f, wih16,            G_ * E_);
  k_cvt_f16 <<<512,  256, 0, stream>>>(Wih_b, wih16 + G_ * E_,  G_ * E_);
  k_cvt_f16 <<<4096, 256, 0, stream>>>(Wl,    wl16,             V_ * K2_);
  k_pack_whh<<<1024, 256, 0, stream>>>(Whh_f, Whh_b, whh_pk);
  k_bias    <<<8,    256, 0, stream>>>(bih_f, bhh_f, bih_b, bhh_b, biasc);

  k_input_gemm<<<dim3(32, 16, 2), 256, 0, stream>>>(emb16, wih16, biasc, pre);
  k_scan      <<<2, 1024, 0, stream>>>(whh_pk, pre, hcat);
  k_final_gemm<<<dim3(M_ / 128, V_ / 64), 256, 0, stream>>>(hcat, wl16, bl, out);
}
